// LSTMModel_58617713656191
// MI455X (gfx1250) — matrix operand order repaired and checked by reading, not compiled, not run on hardware
//
#include <hip/hip_runtime.h>

// ---------------------------------------------------------------------------
// CDNA5 (gfx1250) persistent-LSTM implementation.
//   - bf16 WMMA 16x16x32 with fp32 accumulation (wave32)
//   - weights packed to WMMA B-fragment order once, then stay hot in 192MB L2
//   - activation staging via GLOBAL_LOAD_ASYNC_TO_LDS_B128 (ASYNCcnt path)
//   - one persistent kernel, device-wide barrier per timestep
//   - cell state c lives in registers of the owning wave for all 512 steps
// ---------------------------------------------------------------------------

typedef __bf16 bf16;
typedef __attribute__((ext_vector_type(16))) bf16  bf16x16;
typedef __attribute__((ext_vector_type(8)))  bf16  bf16x8;
typedef __attribute__((ext_vector_type(8)))  float f32x8;

#define LSTM_B   64
#define LSTM_T   512
#define LSTM_I   256
#define LSTM_H   1024
#define G4       4096               // 4*H gate rows
#define K0       (LSTM_I + LSTM_H)  // 1280  (layer0: [x | h])
#define K1       (2 * LSTM_H)       // 2048  (layer1: [h1 | h2])
#define APITCH   (K1 + 8)           // LDS row pitch (bf16 elems), bank-spread
#define NWG      32
#define TPB      256
#define SMEM_BYTES (LSTM_B * APITCH * 2)

// ---------------- device helpers -------------------------------------------

__device__ __forceinline__ float fsig(float x) {
  return 1.0f / (1.0f + __expf(-x));
}
__device__ __forceinline__ float ftanh(float x) {
  float ax = fabsf(x);
  float e  = __expf(2.0f * ax);              // inf-safe: e=inf -> t=1
  float t  = 1.0f - 2.0f / (e + 1.0f);
  return copysignf(t, x);
}

__device__ __forceinline__ f32x8 wmma_bf16(bf16x16 a, bf16x16 b, f32x8 c) {
  // D = A(16x32 bf16) x B(32x16 bf16) + C(16x16 f32)
  return __builtin_amdgcn_wmma_f32_16x16x32_bf16(
      /*neg_a=*/false, a, /*neg_b=*/false, b,
      /*c_mod=*/(short)0, c, /*reuse_a=*/false, /*reuse_b=*/false);
}

// Async 16B copy global -> LDS (no VGPR round-trip, tracked by ASYNCcnt).
// vdst = 32-bit LDS byte offset (relative to wave LDS base), vaddr = 64-bit.
__device__ __forceinline__ void async_copy_b128(unsigned lds_byte_off,
                                                const void* gsrc) {
  asm volatile("global_load_async_to_lds_b128 %0, %1, off"
               :: "v"(lds_byte_off), "v"(gsrc)
               : "memory");
}
__device__ __forceinline__ void wait_async0() {
  asm volatile("s_wait_asynccnt 0x0" ::: "memory");
}

// Device-wide sense barrier (all NWG workgroups resident by construction).
__device__ __forceinline__ void gridsync(unsigned* bar) {
  __syncthreads();
  if (threadIdx.x == 0) {
    unsigned g = __hip_atomic_load(&bar[1], __ATOMIC_RELAXED, __HIP_MEMORY_SCOPE_AGENT);
    unsigned a = __hip_atomic_fetch_add(&bar[0], 1u, __ATOMIC_ACQ_REL, __HIP_MEMORY_SCOPE_AGENT);
    if (a == NWG - 1) {
      __hip_atomic_store(&bar[0], 0u, __ATOMIC_RELAXED, __HIP_MEMORY_SCOPE_AGENT);
      __hip_atomic_store(&bar[1], g + 1u, __ATOMIC_RELEASE, __HIP_MEMORY_SCOPE_AGENT);
    } else {
      while (__hip_atomic_load(&bar[1], __ATOMIC_ACQUIRE, __HIP_MEMORY_SCOPE_AGENT) == g)
        __builtin_amdgcn_s_sleep(2);
    }
  }
  __syncthreads();
}

// ---------------- one-time prep kernels ------------------------------------

__global__ void init_state(unsigned* __restrict__ bar,
                           bf16* __restrict__ h1s0, bf16* __restrict__ h2b0) {
  const int i = blockIdx.x * blockDim.x + threadIdx.x;
  if (i == 0) { bar[0] = 0u; bar[1] = 0u; }
  const int n = LSTM_B * LSTM_H;
  for (int k = i; k < n; k += gridDim.x * blockDim.x) {
    h1s0[k] = (bf16)0.0f;
    h2b0[k] = (bf16)0.0f;
  }
}

__global__ void cvt_x(const float* __restrict__ x, bf16* __restrict__ xb, size_t n) {
  for (size_t i = (size_t)blockIdx.x * blockDim.x + threadIdx.x; i < n;
       i += (size_t)gridDim.x * blockDim.x)
    xb[i] = (bf16)x[i];
}

// Pack W = [Wih | Whh] (fp32, row-major over gate rows n, cols k) into WMMA
// B-fragment order: out[ntile][kt][lane][e] = W[ntile*16 + lane%16]
//                                              [kt*32 + 16*(lane/16) + e]
__global__ void pack_weights(const float* __restrict__ Wih,
                             const float* __restrict__ Whh,
                             int KX, int K, bf16* __restrict__ out) {
  const int NK = K >> 5;
  const size_t total = (size_t)(G4 / 16) * NK * 512;
  for (size_t i = (size_t)blockIdx.x * blockDim.x + threadIdx.x; i < total;
       i += (size_t)gridDim.x * blockDim.x) {
    const int e     = (int)(i & 15);
    const int lane  = (int)((i >> 4) & 31);
    const size_t r  = i >> 9;
    const int kt    = (int)(r % NK);
    const int ntile = (int)(r / NK);
    const int n = ntile * 16 + (lane & 15);
    const int k = kt * 32 + (lane >> 4) * 16 + e;
    const float v = (k < KX) ? Wih[(size_t)n * KX + k]
                             : Whh[(size_t)n * (K - KX) + (k - KX)];
    out[i] = (bf16)v;
  }
}

// ---------------- persistent LSTM kernel -----------------------------------
// Grid: 32 WGs x 256 threads (8 waves). WG b owns nh-tiles {2b, 2b+1}.
// Wave v: m_tile = v>>1, nh = 2b + (v&1); owns all 4 gate tiles of (m,nh).

__global__ __launch_bounds__(TPB, 1) void lstm_persistent(
    const bf16* __restrict__ xbf,     // [B][T][I] bf16
    const bf16* __restrict__ Bpk0,    // packed [256][40][512]
    const bf16* __restrict__ Bpk1,    // packed [256][64][512]
    const float* __restrict__ bias0,  // [4096]
    const float* __restrict__ bias1,  // [4096]
    bf16* __restrict__ h1seq,         // [T+1][B][H], slot0 = zeros
    bf16* __restrict__ h2buf,         // [2][B][H], slot0 = zeros
    unsigned* __restrict__ bar) {
  extern __shared__ bf16 Alds[];      // 64 rows x APITCH

  const int tid   = threadIdx.x;
  const int lane  = tid & 31;
  const int wv    = tid >> 5;
  const int mtile = wv >> 1;
  const int nh    = (int)blockIdx.x * 2 + (wv & 1);
  const int mrow  = lane & 15;
  const int khalf = lane >> 4;        // lane half selects 16-wide K-subblock (contiguous order, same as packed B)
  const unsigned ldsbase = (unsigned)(uintptr_t)(&Alds[0]);

  for (int layer = 0; layer < 2; ++layer) {
    const int K  = layer ? K1 : K0;
    const int KX = layer ? LSTM_H : LSTM_I;
    const int NK = K >> 5;
    const bf16*  __restrict__ Bpk  = layer ? Bpk1 : Bpk0;
    const float* __restrict__ bias = layer ? bias1 : bias0;

    f32x8 cst = {0.f, 0.f, 0.f, 0.f, 0.f, 0.f, 0.f, 0.f};  // cell state tile

    for (int t = 0; t < LSTM_T; ++t) {
      // ---- stage A = [x_t | h_prev] (64 x K bf16) into LDS, async path ----
      const bf16* xs; size_t xstride; const bf16* hs;
      if (layer == 0) {
        xs = xbf + (size_t)t * LSTM_I;          xstride = (size_t)LSTM_T * LSTM_I;
        hs = h1seq + (size_t)t * (LSTM_B * LSTM_H);
      } else {
        xs = h1seq + (size_t)(t + 1) * (LSTM_B * LSTM_H);  xstride = LSTM_H;
        hs = h2buf + (size_t)(t & 1) * (LSTM_B * LSTM_H);
      }
      const int CH  = K >> 3;     // 16B chunks per row (exact multiple of TPB total)
      const int CHX = KX >> 3;
      for (int idx = tid; idx < LSTM_B * CH; idx += TPB) {
        const int r = idx / CH;
        const int c = idx - r * CH;
        const int k = c << 3;
        const bf16* src = (c < CHX) ? (xs + (size_t)r * xstride + k)
                                    : (hs + (size_t)r * LSTM_H + (k - KX));
        const unsigned dst = ldsbase + (unsigned)((r * APITCH + k) * 2);
        async_copy_b128(dst, src);
      }
      wait_async0();
      __syncthreads();

      // ---- GEMM: 4 gate tiles, K-loop of 32 ----
      f32x8 acc0 = {}, acc1 = {}, acc2 = {}, acc3 = {};
      const bf16* arow = &Alds[(size_t)(mtile * 16 + mrow) * APITCH + khalf * 16];
      const bf16* bp0 = Bpk + (size_t)((0 * 64 + nh) * NK) * 512 + lane * 16;
      const bf16* bp1 = Bpk + (size_t)((1 * 64 + nh) * NK) * 512 + lane * 16;
      const bf16* bp2 = Bpk + (size_t)((2 * 64 + nh) * NK) * 512 + lane * 16;
      const bf16* bp3 = Bpk + (size_t)((3 * 64 + nh) * NK) * 512 + lane * 16;

      for (int kt = 0; kt < NK; ++kt) {
        // A fragment (contiguous, matches packed B): e0..15 -> k=16*half+e
        bf16x8 alo = *(const bf16x8*)(arow + kt * 32);
        bf16x8 ahi = *(const bf16x8*)(arow + kt * 32 + 8);
        bf16x16 af = __builtin_shufflevector(alo, ahi,
            0, 1, 2, 3, 4, 5, 6, 7, 8, 9, 10, 11, 12, 13, 14, 15);
        bf16x16 b0f = *(const bf16x16*)(bp0 + (size_t)kt * 512);
        bf16x16 b1f = *(const bf16x16*)(bp1 + (size_t)kt * 512);
        bf16x16 b2f = *(const bf16x16*)(bp2 + (size_t)kt * 512);
        bf16x16 b3f = *(const bf16x16*)(bp3 + (size_t)kt * 512);
        acc0 = wmma_bf16(af, b0f, acc0);
        acc1 = wmma_bf16(af, b1f, acc1);
        acc2 = wmma_bf16(af, b2f, acc2);
        acc3 = wmma_bf16(af, b3f, acc3);
      }

      // ---- pointwise LSTM cell (all in registers) ----
      const int ncol = nh * 16 + (lane & 15);
      const float bi = bias[0 * LSTM_H + ncol];
      const float bf = bias[1 * LSTM_H + ncol];
      const float bg = bias[2 * LSTM_H + ncol];
      const float bo = bias[3 * LSTM_H + ncol];
      bf16* hdst = (layer == 0)
          ? (h1seq + (size_t)(t + 1) * (LSTM_B * LSTM_H))
          : (h2buf + (size_t)((t + 1) & 1) * (LSTM_B * LSTM_H));
#pragma unroll
      for (int j = 0; j < 8; ++j) {   // D-layout: m = mtile*16 + 8*half + j
        const float iv = fsig(acc0[j] + bi);
        const float fv = fsig(acc1[j] + bf);
        const float gv = ftanh(acc2[j] + bg);
        const float ov = fsig(acc3[j] + bo);
        const float cv = fv * cst[j] + iv * gv;
        cst[j] = cv;
        const float hv = ov * ftanh(cv);
        const int m = mtile * 16 + khalf * 8 + j;
        hdst[(size_t)m * LSTM_H + ncol] = (bf16)hv;
      }
      gridsync(bar);   // publish h_t to every WG before step t+1
    }
  }
}

// ---------------- final FC + ReLU ------------------------------------------
// out[64][1000] = relu(h2_last @ Wfc^T + bfc). Block = (m, 256-class slice).
__global__ void fc_relu(const bf16* __restrict__ h2, const float* __restrict__ Wfc,
                        const float* __restrict__ bfc, float* __restrict__ out) {
  __shared__ float hrow[LSTM_H];
  const int m = blockIdx.x >> 2;
  const int cbase = (blockIdx.x & 3) * 256;
  for (int i = threadIdx.x; i < LSTM_H; i += blockDim.x)
    hrow[i] = (float)h2[(size_t)m * LSTM_H + i];
  __syncthreads();
  const int c = cbase + threadIdx.x;
  if (c < 1000) {
    const float* w = Wfc + (size_t)c * LSTM_H;
    float acc = bfc[c];
    for (int k = 0; k < LSTM_H; ++k) acc = fmaf(hrow[k], w[k], acc);
    out[(size_t)m * 1000 + c] = fmaxf(acc, 0.0f);
  }
}

// ---------------- host entry ------------------------------------------------

extern "C" void kernel_launch(void* const* d_in, const int* in_sizes, int n_in,
                              void* d_out, int out_size, void* d_ws, size_t ws_size,
                              hipStream_t stream) {
  (void)in_sizes; (void)n_in; (void)out_size; (void)ws_size;
  const float* x    = (const float*)d_in[0];
  const float* Wih0 = (const float*)d_in[1];
  const float* Whh0 = (const float*)d_in[2];
  const float* b0   = (const float*)d_in[3];
  const float* Wih1 = (const float*)d_in[4];
  const float* Whh1 = (const float*)d_in[5];
  const float* b1   = (const float*)d_in[6];
  const float* Wfc  = (const float*)d_in[7];
  const float* bfc  = (const float*)d_in[8];
  float* out = (float*)d_out;

  char* ws = (char*)d_ws;
  size_t off = 0;
  auto carve = [&](size_t bytes) {
    char* p = ws + off;
    off = (off + bytes + 255) & ~(size_t)255;
    return p;
  };
  unsigned* bar = (unsigned*)carve(256);
  bf16* xbf   = (bf16*)carve((size_t)LSTM_B * LSTM_T * LSTM_I * 2);        // 16.8 MB
  bf16* h1seq = (bf16*)carve((size_t)(LSTM_T + 1) * LSTM_B * LSTM_H * 2);  // 67.2 MB
  bf16* h2buf = (bf16*)carve((size_t)2 * LSTM_B * LSTM_H * 2);             // 256 KB
  bf16* Bpk0  = (bf16*)carve((size_t)G4 * K0 * 2);                         // 10.5 MB
  bf16* Bpk1  = (bf16*)carve((size_t)G4 * K1 * 2);                         // 16.8 MB

  init_state<<<256, 256, 0, stream>>>(bar, h1seq, h2buf);
  cvt_x<<<2048, 256, 0, stream>>>(x, xbf, (size_t)LSTM_B * LSTM_T * LSTM_I);
  pack_weights<<<2048, 256, 0, stream>>>(Wih0, Whh0, LSTM_I, K0, Bpk0);
  pack_weights<<<2048, 256, 0, stream>>>(Wih1, Whh1, LSTM_H, K1, Bpk1);
  lstm_persistent<<<NWG, TPB, SMEM_BYTES, stream>>>(xbf, Bpk0, Bpk1, b0, b1,
                                                    h1seq, h2buf, bar);
  fc_relu<<<LSTM_B * 4, 256, 0, stream>>>(h2buf, Wfc, bfc, out);
}
